// GridPooling_34265249088351
// MI455X (gfx1250) — matrix operand order repaired and checked by reading, not compiled, not run on hardware
//
#include <hip/hip_runtime.h>
#include <hip/hip_bf16.h>
#include <stdint.h>

#define N_PTS   200000
#define C_IN    128
#define C_OUT   256
#define GRIDSZ  0.05f
#define BN_EPS  1e-5f
#define CAP_BINS (1 << 22)   // occupancy bitmap capacity (real: ~32K bins)
#define CAP_VOX  65536       // max unique voxels (real: <= 32000)
#define MAX_B    16

typedef __attribute__((ext_vector_type(16))) _Float16 v16h;
typedef __attribute__((ext_vector_type(8)))  float    v8f;

// ---- monotone float<->uint mapping for atomic min/max on floats ----
__device__ __forceinline__ unsigned mapf(float f) {
  unsigned u = __float_as_uint(f);
  return (u & 0x80000000u) ? ~u : (u | 0x80000000u);
}
__device__ __forceinline__ float unmapf(unsigned u) {
  return (u & 0x80000000u) ? __uint_as_float(u & 0x7fffffffu) : __uint_as_float(~u);
}

__device__ __forceinline__ int point_batch(int i, const int* __restrict__ offs, int nb) {
  int b = 0;
  for (int j = 0; j < nb - 1; ++j) b += (i >= offs[j]) ? 1 : 0;
  return b;
}

// ================= voxel pipeline =================

__global__ void k_start_min(const float* __restrict__ xyz, const int* __restrict__ offs,
                            int nb, unsigned* __restrict__ startM) {
  int i = blockIdx.x * blockDim.x + threadIdx.x;
  if (i >= N_PTS) return;
  int b = point_batch(i, offs, nb);
  #pragma unroll
  for (int j = 0; j < 3; ++j) atomicMin(&startM[b * 3 + j], mapf(xyz[i * 3 + j]));
}

__global__ void k_dims(const float* __restrict__ xyz, const int* __restrict__ offs,
                       int nb, const unsigned* __restrict__ startM, int* __restrict__ dims) {
  int i = blockIdx.x * blockDim.x + threadIdx.x;
  if (i >= N_PTS) return;
  int b = point_batch(i, offs, nb);
  #pragma unroll
  for (int j = 0; j < 3; ++j) {
    int c = (int)floorf((xyz[i * 3 + j] - unmapf(startM[b * 3 + j])) / GRIDSZ);
    atomicMax(&dims[j], c + 1);
  }
}

__global__ void k_lin(const float* __restrict__ xyz, const int* __restrict__ offs, int nb,
                      const unsigned* __restrict__ startM, const int* __restrict__ dims,
                      int* __restrict__ linArr, unsigned* __restrict__ occ) {
  int i = blockIdx.x * blockDim.x + threadIdx.x;
  if (i >= N_PTS) return;
  int b = point_batch(i, offs, nb);
  int c[3];
  #pragma unroll
  for (int j = 0; j < 3; ++j)
    c[j] = (int)floorf((xyz[i * 3 + j] - unmapf(startM[b * 3 + j])) / GRIDSZ);
  long lin = (((long)b * dims[0] + c[0]) * dims[1] + c[1]) * dims[2] + c[2];
  int l = (lin < 0) ? 0 : ((lin >= CAP_BINS) ? (CAP_BINS - 1) : (int)lin);
  linArr[i] = l;
  occ[l] = 1u;
}

// Single-block exclusive scan over the bin space -> sorted-unique ranks.
__global__ void k_scan(const unsigned* __restrict__ occ, unsigned* __restrict__ rank,
                       const int* __restrict__ dims, int nb,
                       int* __restrict__ Mptr, int* __restrict__ batchCnt) {
  __shared__ unsigned s[1024];
  __shared__ unsigned carry;
  int tid = threadIdx.x;
  if (tid == 0) carry = 0;
  __syncthreads();
  int vol = dims[0] * dims[1] * dims[2];
  long totL = (long)nb * vol;
  int total = (totL > CAP_BINS) ? CAP_BINS : (int)totL;
  for (int base = 0; base < total; base += 1024) {
    int i = base + tid;
    unsigned v = (i < total) ? occ[i] : 0u;
    s[tid] = v;
    __syncthreads();
    for (int offn = 1; offn < 1024; offn <<= 1) {
      unsigned t = (tid >= offn) ? s[tid - offn] : 0u;
      __syncthreads();
      if (tid >= offn) s[tid] += t;
      __syncthreads();
    }
    unsigned cbase = carry;
    unsigned chunk = s[1023];
    __syncthreads();
    if (tid == 0) carry = cbase + chunk;
    if (i < total) {
      rank[i] = cbase + (s[tid] - v);   // exclusive
      if (v) {
        int b = i / vol;
        atomicAdd(&batchCnt[b], 1);
      }
    }
    __syncthreads();
  }
  if (tid == 0) *Mptr = (int)carry;
}

// inv lookup + write inv to d_out + accumulate xyz segment-sum/counts
__global__ void k_inv(const float* __restrict__ xyz, const int* __restrict__ linArr,
                      const unsigned* __restrict__ rank, const int* __restrict__ Mptr,
                      int* __restrict__ invArr, float* __restrict__ voxSum,
                      unsigned* __restrict__ voxCnt, int* __restrict__ out_i) {
  int i = blockIdx.x * blockDim.x + threadIdx.x;
  if (i >= N_PTS) return;
  int v = (int)rank[linArr[i]];
  invArr[i] = v;
  int M = *Mptr;
  size_t invBase = (size_t)3 * M + (size_t)C_OUT * M + 4;
  out_i[invBase + i] = v;
  if ((unsigned)v < CAP_VOX) {
    atomicAdd(&voxCnt[v], 1u);
    #pragma unroll
    for (int j = 0; j < 3; ++j) atomicAdd(&voxSum[v * 3 + j], xyz[i * 3 + j]);
  }
}

// ================= WMMA GEMMs =================
// One wave owns one 16x16 output tile. 8 waves/block, 25000 blocks = 200000 tiles.

__global__ void __launch_bounds__(256)
k_gemm1(const float* __restrict__ X, const float* __restrict__ W1,
        const float* __restrict__ b1, float* __restrict__ H) {
  int lane = threadIdx.x & 31;
  int wave = threadIdx.x >> 5;
  int tile = blockIdx.x * 8 + wave;
  int row0 = (tile >> 4) * 16;
  int col0 = (tile & 15) * 16;
  int half = lane >> 4;          // 0 or 1
  int mn   = lane & 15;          // A-row / B-col within tile
  int kbA  = half * 8;           // A half-lane K base
  int kbB  = half * 8;           // B half-lane K base
  v8f acc = {};
  for (int ks = 0; ks < 4; ++ks) {           // K = 128 in steps of 32
    int k0 = ks * 32;
    v16h A, B;
    {
      const float4* p = (const float4*)(X + (size_t)(row0 + mn) * C_IN + k0 + kbA);
      float4 t0 = p[0], t1 = p[1], t2 = p[4], t3 = p[5];   // K: +0..7 and +16..23
      float f[16] = {t0.x,t0.y,t0.z,t0.w, t1.x,t1.y,t1.z,t1.w,
                     t2.x,t2.y,t2.z,t2.w, t3.x,t3.y,t3.z,t3.w};
      #pragma unroll
      for (int e = 0; e < 16; ++e) A[e] = (_Float16)f[e];
    }
    {
      const float4* p = (const float4*)(W1 + (size_t)(col0 + mn) * C_IN + k0 + kbB);
      float4 t0 = p[0], t1 = p[1], t2 = p[4], t3 = p[5];   // K: +0..7 and +16..23
      float f[16] = {t0.x,t0.y,t0.z,t0.w, t1.x,t1.y,t1.z,t1.w,
                     t2.x,t2.y,t2.z,t2.w, t3.x,t3.y,t3.z,t3.w};
      #pragma unroll
      for (int e = 0; e < 16; ++e) B[e] = (_Float16)f[e];
    }
    acc = __builtin_amdgcn_wmma_f32_16x16x32_f16(false, A, false, B, (short)0, acc, false, false);
  }
  int mrow = half * 8;
  int c = col0 + mn;
  float bias = b1[c];
  #pragma unroll
  for (int v = 0; v < 8; ++v)
    H[(size_t)(row0 + mrow + v) * C_OUT + c] = acc[v] + bias;
}

__global__ void k_bn_stats(const float* __restrict__ H, float* __restrict__ chanSum,
                           float* __restrict__ chanSq) {
  int c = threadIdx.x;                                  // 256 channels
  int rpb = (N_PTS + gridDim.x - 1) / gridDim.x;
  int r0 = blockIdx.x * rpb;
  int r1 = r0 + rpb; if (r1 > N_PTS) r1 = N_PTS;
  float s = 0.f, q = 0.f;
  for (int r = r0; r < r1; ++r) {
    float v = H[(size_t)r * C_OUT + c];
    s += v; q += v * v;
  }
  atomicAdd(&chanSum[c], s);
  atomicAdd(&chanSq[c], q);
}

__global__ void k_bn_finalize(const float* __restrict__ chanSum, const float* __restrict__ chanSq,
                              const float* __restrict__ gamma, const float* __restrict__ beta,
                              float* __restrict__ scale, float* __restrict__ shift) {
  int c = threadIdx.x;
  float mu  = chanSum[c] / (float)N_PTS;
  float var = chanSq[c] / (float)N_PTS - mu * mu;       // biased, matches jnp.var
  float sc  = gamma[c] * rsqrtf(var + BN_EPS);
  scale[c] = sc;
  shift[c] = beta[c] - mu * sc;
}

// GEMM2 with BN+ReLU on A-fragments, fused segment-max scatter epilogue.
__global__ void __launch_bounds__(256)
k_gemm2(const float* __restrict__ H, const float* __restrict__ W2,
        const float* __restrict__ b2, const float* __restrict__ scale,
        const float* __restrict__ shift, const int* __restrict__ invArr,
        unsigned* __restrict__ featMax) {
  __shared__ float s_scale[C_OUT], s_shift[C_OUT];
  s_scale[threadIdx.x] = scale[threadIdx.x];
  s_shift[threadIdx.x] = shift[threadIdx.x];
  __syncthreads();
  int lane = threadIdx.x & 31;
  int wave = threadIdx.x >> 5;
  int tile = blockIdx.x * 8 + wave;
  int row0 = (tile >> 4) * 16;
  int col0 = (tile & 15) * 16;
  int half = lane >> 4;
  int mn   = lane & 15;
  int kbA  = half * 8;
  int kbB  = half * 8;
  v8f acc = {};
  for (int ks = 0; ks < 8; ++ks) {           // K = 256 in steps of 32
    int k0 = ks * 32;
    v16h A, B;
    {
      const float4* p = (const float4*)(H + (size_t)(row0 + mn) * C_OUT + k0 + kbA);
      float4 t0 = p[0], t1 = p[1], t2 = p[4], t3 = p[5];
      float f[16] = {t0.x,t0.y,t0.z,t0.w, t1.x,t1.y,t1.z,t1.w,
                     t2.x,t2.y,t2.z,t2.w, t3.x,t3.y,t3.z,t3.w};
      #pragma unroll
      for (int e = 0; e < 16; ++e) {
        int k = k0 + kbA + (e < 8 ? e : e + 8);       // element -> channel index
        float y = f[e] * s_scale[k] + s_shift[k];
        A[e] = (_Float16)(y > 0.f ? y : 0.f);         // ReLU
      }
    }
    {
      const float4* p = (const float4*)(W2 + (size_t)(col0 + mn) * C_OUT + k0 + kbB);
      float4 t0 = p[0], t1 = p[1], t2 = p[4], t3 = p[5];
      float f[16] = {t0.x,t0.y,t0.z,t0.w, t1.x,t1.y,t1.z,t1.w,
                     t2.x,t2.y,t2.z,t2.w, t3.x,t3.y,t3.z,t3.w};
      #pragma unroll
      for (int e = 0; e < 16; ++e) B[e] = (_Float16)f[e];
    }
    acc = __builtin_amdgcn_wmma_f32_16x16x32_f16(false, A, false, B, (short)0, acc, false, false);
  }
  int mrow = half * 8;
  int c = col0 + mn;
  float bias = b2[c];
  #pragma unroll
  for (int v = 0; v < 8; ++v) {
    float val = acc[v] + bias;
    int r = row0 + mrow + v;
    int vox = invArr[r];
    if ((unsigned)vox < CAP_VOX)
      atomicMax(&featMax[(size_t)vox * C_OUT + c], mapf(val));
  }
}

// ================= finalize into d_out =================

__global__ void k_fin_xyz(const float* __restrict__ voxSum, const unsigned* __restrict__ voxCnt,
                          const int* __restrict__ Mptr, float* __restrict__ out_f) {
  int idx = blockIdx.x * blockDim.x + threadIdx.x;
  int M = *Mptr;
  if (idx >= M * 3) return;
  out_f[idx] = voxSum[idx] / (float)voxCnt[idx / 3];
}

__global__ void k_fin_feat(const unsigned* __restrict__ featMax, const int* __restrict__ Mptr,
                           float* __restrict__ out_f) {
  long idx = (long)blockIdx.x * blockDim.x + threadIdx.x;
  int M = *Mptr;
  if (idx >= (long)M * C_OUT) return;
  out_f[(size_t)3 * M + idx] = unmapf(featMax[idx]);
}

__global__ void k_fin_offset(const int* __restrict__ batchCnt, const int* __restrict__ Mptr,
                             int nb, int* __restrict__ out_i) {
  if (threadIdx.x == 0 && blockIdx.x == 0) {
    int M = *Mptr;
    size_t base = (size_t)3 * M + (size_t)C_OUT * M;
    int acc = 0;
    for (int b = 0; b < nb; ++b) { acc += batchCnt[b]; out_i[base + b] = acc; }
  }
}

// ================= host side =================

extern "C" void kernel_launch(void* const* d_in, const int* in_sizes, int n_in,
                              void* d_out, int out_size, void* d_ws, size_t ws_size,
                              hipStream_t stream) {
  const float* xyz   = (const float*)d_in[0];
  const float* feats = (const float*)d_in[1];
  const int*   offs  = (const int*)d_in[2];
  const float* W1    = (const float*)d_in[3];
  const float* b1    = (const float*)d_in[4];
  const float* gamma = (const float*)d_in[5];
  const float* beta  = (const float*)d_in[6];
  const float* W2    = (const float*)d_in[7];
  const float* b2    = (const float*)d_in[8];
  int nb = in_sizes[2];
  if (nb > MAX_B) nb = MAX_B;

  char* ws = (char*)d_ws;
  size_t wsoff = 0;
  auto alloc = [&](size_t bytes) -> void* {
    void* p = ws + wsoff;
    wsoff = (wsoff + bytes + 255) & ~(size_t)255;
    return p;
  };
  unsigned* startM   = (unsigned*)alloc(MAX_B * 3 * 4);
  int*      dims     = (int*)     alloc(3 * 4);
  int*      Mptr     = (int*)     alloc(4);
  int*      batchCnt = (int*)     alloc(MAX_B * 4);
  float*    chanSum  = (float*)   alloc(C_OUT * 4);
  float*    chanSq   = (float*)   alloc(C_OUT * 4);
  float*    scaleB   = (float*)   alloc(C_OUT * 4);
  float*    shiftB   = (float*)   alloc(C_OUT * 4);
  int*      linArr   = (int*)     alloc((size_t)N_PTS * 4);
  int*      invArr   = (int*)     alloc((size_t)N_PTS * 4);
  unsigned* occ      = (unsigned*)alloc((size_t)CAP_BINS * 4);
  unsigned* rank     = (unsigned*)alloc((size_t)CAP_BINS * 4);
  float*    voxSum   = (float*)   alloc((size_t)CAP_VOX * 3 * 4);
  unsigned* voxCnt   = (unsigned*)alloc((size_t)CAP_VOX * 4);
  unsigned* featMax  = (unsigned*)alloc((size_t)CAP_VOX * C_OUT * 4);
  float*    H        = (float*)   alloc((size_t)N_PTS * C_OUT * 4);

  // per-call re-init (graph-capture-safe memsets)
  hipMemsetAsync(startM, 0xFF, MAX_B * 3 * 4, stream);       // +inf in ordered-uint space
  hipMemsetAsync(dims, 0, 3 * 4, stream);
  hipMemsetAsync(Mptr, 0, 4, stream);
  hipMemsetAsync(batchCnt, 0, MAX_B * 4, stream);
  hipMemsetAsync(chanSum, 0, C_OUT * 4, stream);
  hipMemsetAsync(chanSq, 0, C_OUT * 4, stream);
  hipMemsetAsync(occ, 0, (size_t)CAP_BINS * 4, stream);
  hipMemsetAsync(voxSum, 0, (size_t)CAP_VOX * 3 * 4, stream);
  hipMemsetAsync(voxCnt, 0, (size_t)CAP_VOX * 4, stream);
  hipMemsetAsync(featMax, 0, (size_t)CAP_VOX * C_OUT * 4, stream); // 0 < map(-inf)

  int pb = (N_PTS + 255) / 256;
  k_start_min<<<pb, 256, 0, stream>>>(xyz, offs, nb, startM);
  k_dims     <<<pb, 256, 0, stream>>>(xyz, offs, nb, startM, dims);
  k_lin      <<<pb, 256, 0, stream>>>(xyz, offs, nb, startM, dims, linArr, occ);
  k_scan     <<<1, 1024, 0, stream>>>(occ, rank, dims, nb, Mptr, batchCnt);
  k_inv      <<<pb, 256, 0, stream>>>(xyz, linArr, rank, Mptr, invArr, voxSum, voxCnt,
                                      (int*)d_out);

  int gemmBlocks = (N_PTS / 16) * (C_OUT / 16) / 8;   // 25000
  k_gemm1<<<gemmBlocks, 256, 0, stream>>>(feats, W1, b1, H);
  k_bn_stats<<<256, 256, 0, stream>>>(H, chanSum, chanSq);
  k_bn_finalize<<<1, 256, 0, stream>>>(chanSum, chanSq, gamma, beta, scaleB, shiftB);
  k_gemm2<<<gemmBlocks, 256, 0, stream>>>(H, W2, b2, scaleB, shiftB, invArr, featMax);

  k_fin_xyz <<<(CAP_VOX * 3 + 255) / 256, 256, 0, stream>>>(voxSum, voxCnt, Mptr, (float*)d_out);
  k_fin_feat<<<CAP_VOX, 256, 0, stream>>>(featMax, Mptr, (float*)d_out);
  k_fin_offset<<<1, 64, 0, stream>>>(batchCnt, Mptr, nb, (int*)d_out);
}
